// MultiHeadAttention_3633542332589
// MI455X (gfx1250) — matrix operand order repaired and checked by reading, not compiled, not run on hardware
//
#include <hip/hip_runtime.h>

// ---------------------------------------------------------------------------
// MI455X (gfx1250, wave32) multi-head attention.
// All matmuls (Q/KV/out projections, QK^T, PV, and softmax row-sums) run on
// v_wmma_f32_16x16x32_f16. K-tile staging uses the Tensor Data Mover when the
// toolchain exposes the builtin (s_wait_tensorcnt sync), else manual staging.
// ---------------------------------------------------------------------------

typedef _Float16 half_t;
typedef __attribute__((ext_vector_type(16))) _Float16 v16h;
typedef __attribute__((ext_vector_type(8)))  _Float16 h8;
typedef __attribute__((ext_vector_type(2)))  _Float16 h2;
typedef __attribute__((ext_vector_type(8)))  float    v8f;
typedef __attribute__((ext_vector_type(4)))  float    f4;
typedef __attribute__((ext_vector_type(4)))  unsigned int v4u;
typedef __attribute__((ext_vector_type(8)))  int      v8i;
typedef __attribute__((ext_vector_type(4)))  int      v4i;

#if defined(__gfx1250__) && __has_builtin(__builtin_amdgcn_tensor_load_to_lds)
  #define ATTN_USE_TDM 1
  #if __has_include(<hip/amd_detail/amd_gfx1250_TDM.h>)
    #define ATTN_TDM_6ARG 1
  #endif
#endif

__device__ __forceinline__ v8f wmma_f16(v16h a, v16h b, v8f c) {
  // D = A(16x32 f16) * B(32x16 f16) + C(16x16 f32)
  return __builtin_amdgcn_wmma_f32_16x16x32_f16(
      /*neg_a=*/false, a, /*neg_b=*/false, b,
      /*c_mod=*/(short)0, c, /*reuse_a=*/false, /*reuse_b=*/false);
}

// A-fragment (16x32 f16): lane = M (lane&15); per ISA table element i=2v+pos
// maps to K = (v>=4?16:0) + grp*8 + (v&3)*2 + pos.
__device__ __forceinline__ v16h fragA_ld(const half_t* __restrict__ tile,
                                         int row_off, int ldh, int kbase) {
  const int lane = threadIdx.x & 31;
  const int row  = row_off + (lane & 15);
  const int grp  = lane >> 4;
  const half_t* base = tile + row * ldh + kbase + grp * 8;
  v16h f;
#pragma unroll
  for (int v = 0; v < 8; ++v) {
    const int off = ((v & 4) ? 16 : 0) + (v & 3) * 2;  // halves
    h2 x = *(const h2*)(base + off);
    f[2 * v]     = x.x;
    f[2 * v + 1] = x.y;
  }
  return f;
}

// B-fragment (32x16 f16): lane = N (lane&15); per lane K = grp*8 + 0..7 and
// 16 + grp*8 + 0..7 => two 16B reads from a row-major [N][K] tile.
__device__ __forceinline__ v16h fragB_ld(const half_t* __restrict__ tile,
                                         int col_off, int ldh, int kbase) {
  const int lane = threadIdx.x & 31;
  const int n    = col_off + (lane & 15);
  const int grp  = lane >> 4;
  const half_t* base = tile + n * ldh + kbase + grp * 8;
  h8 lo = *(const h8*)(base);
  h8 hi = *(const h8*)(base + 16);
  v16h f;
#pragma unroll
  for (int i = 0; i < 8; ++i) { f[i] = lo[i]; f[8 + i] = hi[i]; }
  return f;
}

// 8-element staging loaders (fused f32->f16 conversion when A is fp32).
__device__ __forceinline__ h8 load8h(const float* __restrict__ p) {
  f4 a = *(const f4*)p;
  f4 b = *(const f4*)(p + 4);
  h8 r;
  r[0] = (half_t)a.x; r[1] = (half_t)a.y; r[2] = (half_t)a.z; r[3] = (half_t)a.w;
  r[4] = (half_t)b.x; r[5] = (half_t)b.y; r[6] = (half_t)b.z; r[7] = (half_t)b.w;
  return r;
}
__device__ __forceinline__ h8 load8h(const half_t* __restrict__ p) {
  return *(const h8*)p;
}

// ---------------------------------------------------------------------------
// GEMM: C[M,N] = alpha * (A[M,K] @ W[N,K]^T + bias[N])
// Block tile 128(M) x 128(N), K-step 64, 256 threads = 8 waves; each wave owns
// a 16x128 C strip => 16 WMMA per K-step between barriers.
// ---------------------------------------------------------------------------
template <typename AT, bool OUT_F32>
__global__ __launch_bounds__(256)
void gemm_bias_kernel(const AT* __restrict__ A, const float* __restrict__ W,
                      const float* __restrict__ bias, void* __restrict__ Cout,
                      int M, int N, int K, float alpha) {
  constexpr int LDT = 72;  // halves: 64 + 8 pad (144B rows; 16B aligned, banks spread)
  __shared__ __attribute__((aligned(16))) half_t As[128 * LDT];
  __shared__ __attribute__((aligned(16))) half_t Ws[128 * LDT];

  const int tid     = threadIdx.x;
  const int wave    = tid >> 5;
  const int lane    = tid & 31;
  const int grp     = lane >> 4;
  const int cl      = lane & 15;
  const int rowBase = blockIdx.y * 128;
  const int colBase = blockIdx.x * 128;

  v8f acc[8] = {};

  for (int k0 = 0; k0 < K; k0 += 64) {
    // Stage A and W tiles: 128 rows x 64 halves each; 4 passes x 16B/thread.
#pragma unroll
    for (int p = 0; p < 4; ++p) {
      const int idx = p * 256 + tid;
      const int r   = idx >> 3;
      const int cg  = idx & 7;
      const AT* srcA = A + (size_t)(rowBase + r) * K + k0 + cg * 8;
      *(h8*)(&As[r * LDT + cg * 8]) = load8h(srcA);
      const float* srcW = W + (size_t)(colBase + r) * K + k0 + cg * 8;
      *(h8*)(&Ws[r * LDT + cg * 8]) = load8h(srcW);
    }
    __syncthreads();

    const v16h a0 = fragA_ld(As, wave * 16, LDT, 0);
    const v16h a1 = fragA_ld(As, wave * 16, LDT, 32);
#pragma unroll
    for (int nt = 0; nt < 8; ++nt) {
      acc[nt] = wmma_f16(a0, fragB_ld(Ws, nt * 16, LDT, 0), acc[nt]);
      acc[nt] = wmma_f16(a1, fragB_ld(Ws, nt * 16, LDT, 32), acc[nt]);
    }
    __syncthreads();
  }

  // Epilogue: bias + alpha.
#pragma unroll
  for (int nt = 0; nt < 8; ++nt) {
    const int n   = colBase + nt * 16 + cl;
    const float b = bias[n];
#pragma unroll
    for (int r = 0; r < 8; ++r) {
      const int m = rowBase + wave * 16 + r + 8 * grp;
      const float val = alpha * (acc[nt][r] + b);
      if (OUT_F32)
        ((float*)Cout)[(size_t)m * N + n] = val;
      else
        ((half_t*)Cout)[(size_t)m * N + n] = (half_t)val;
    }
  }
}

// ---------------------------------------------------------------------------
// Flash attention: one (b,h) and 128 query rows per block; 8 waves each own a
// 16-row strip. 64-key tiles; K staged by the Tensor Data Mover (LDS-padded
// via D# pad fields) when available; V staged manually transposed. Softmax
// row-sums computed on the matrix pipe as P @ ones (extra WMMA accumulator),
// leaving only the row-max as a cross-lane (ds_bpermute) reduction.
// ---------------------------------------------------------------------------
__global__ __launch_bounds__(256)
void flash_attn_kernel(const half_t* __restrict__ qf,   // (B*Nq) x 1024
                       const half_t* __restrict__ kvf,  // (B*N)  x 2048
                       half_t* __restrict__ valf,       // (B*Nq) x 1024
                       int B, int Nq, int N) {
  constexpr int LDK = 72;  // [key 0..63][d 0..63] + 8-half pad
  constexpr int LDV = 72;  // [d 0..63][key 0..63] + 8-half pad (transposed V)
  constexpr int LDP = 72;  // per-wave P tile [m 0..15][key 0..63]

  __shared__ __attribute__((aligned(16))) half_t Kt[64 * LDK];
  __shared__ __attribute__((aligned(16))) half_t Vt[64 * LDV];
  __shared__ __attribute__((aligned(16))) half_t Pt[8 * 16 * LDP];

  const int bh    = blockIdx.y;
  const int b     = bh >> 4;   // H = 16
  const int h     = bh & 15;
  const int qBase = blockIdx.x * 128;
  const int tid   = threadIdx.x;
  const int wave  = tid >> 5;
  const int lane  = tid & 31;
  const int grp   = lane >> 4;
  const int cl    = lane & 15;

  half_t* Pw = &Pt[wave * 16 * LDP];

  // All-ones B fragment: row-sum of P via the matrix pipe.
  v16h vones;
#pragma unroll
  for (int i = 0; i < 16; ++i) vones[i] = (half_t)1.0f;

  // Query fragments (q pre-scaled by 1/sqrt(Dh) in the projection).
  const half_t* qrow = qf + (size_t)(b * Nq + qBase + wave * 16) * 1024 + h * 64;
  const v16h qa0 = fragA_ld(qrow, 0, 1024, 0);
  const v16h qa1 = fragA_ld(qrow, 0, 1024, 32);

  float mrow[8];
  v8f accl = {};    // running softmax denominator (per row, lane-replicated)
  v8f acc[4] = {};  // output accumulator (16 x 64)
#pragma unroll
  for (int r = 0; r < 8; ++r) mrow[r] = -1e30f;

  for (int kt = 0; kt < N; kt += 64) {
#if defined(ATTN_USE_TDM)
    // --- K tile via Tensor Data Mover: 64x64-half 2D tile, row stride 2048
    // halves, LDS padding 4 DWORDs every 32 DWORDs => LDS row stride 72. ---
    if (wave == 0) {
      const unsigned long long ga =
          (unsigned long long)(size_t)(kvf + (size_t)(b * N + kt) * 2048 + h * 128);
      v4u g0;
      g0[0] = 1u;                                     // count = 1
      g0[1] = (unsigned int)(size_t)(&Kt[0]);         // LDS byte address
      g0[2] = (unsigned int)ga;                       // global addr [31:0]
      g0[3] = (unsigned int)((ga >> 32) & 0x01FFFFFFu) | (2u << 30);  // type=2
      v8i g1;
      g1[0] = (int)((1u << 16) |   // data_size = 2 bytes
                    (1u << 20) |   // pad_enable
                    (4u << 22) |   // pad_interval: every 32 DWORDs
                    (3u << 25));   // pad_amount: 4 DWORDs (8 halves)
      g1[1] = (int)(64u << 16);    // tensor_dim0 = 64 (lo16)
      g1[2] = (int)(64u << 16);    // tensor_dim0 hi | tensor_dim1 = 64 (lo16)
      g1[3] = (int)(64u << 16);    // tensor_dim1 hi | tile_dim0 = 64
      g1[4] = 64;                  // tile_dim1 = 64, tile_dim2 = 0
      g1[5] = 2048;                // tensor_dim0_stride (lo32, halves)
      g1[6] = 0;
      g1[7] = 0;
      v4i z4 = {0, 0, 0, 0};
#if defined(ATTN_TDM_6ARG)
      v8i z8 = {0, 0, 0, 0, 0, 0, 0, 0};
      __builtin_amdgcn_tensor_load_to_lds(g0, g1, z4, z4, z8, 0);
#else
      __builtin_amdgcn_tensor_load_to_lds(g0, g1, z4, z4, 0);
#endif
    }
    // V tile staged (transposed) by all waves: 512 items, 2 passes.
#pragma unroll
    for (int p = 0; p < 2; ++p) {
      const int j   = p * 256 + tid;
      const int row = j >> 3;
      const int cg  = j & 7;
      const half_t* src =
          kvf + (size_t)(b * N + kt + row) * 2048 + h * 128 + 64 + cg * 8;
      const h8 d = *(const h8*)src;
#pragma unroll
      for (int e = 0; e < 8; ++e) Vt[(cg * 8 + e) * LDV + row] = d[e];
      if (kt + 64 < N) __builtin_prefetch(src + 64 * 2048, 0, 0);
    }
    if (wave == 0) __builtin_amdgcn_s_wait_tensorcnt(0);
#else
    // --- Manual K + V staging: 1024 items, 4 passes (pass-uniform K/V split).
#pragma unroll
    for (int p = 0; p < 4; ++p) {
      const int idx  = p * 256 + tid;
      const int part = idx >> 9;  // 0 = K (p=0,1), 1 = V (p=2,3)
      const int j    = idx & 511;
      const int row  = j >> 3;
      const int cg   = j & 7;
      const half_t* src =
          kvf + (size_t)(b * N + kt + row) * 2048 + h * 128 + part * 64 + cg * 8;
      const h8 d = *(const h8*)src;
      if (part == 0) {
        *(h8*)(&Kt[row * LDK + cg * 8]) = d;
      } else {
#pragma unroll
        for (int e = 0; e < 8; ++e) Vt[(cg * 8 + e) * LDV + row] = d[e];
      }
      if (kt + 64 < N) __builtin_prefetch(src + 64 * 2048, 0, 0);
    }
#endif
    __syncthreads();

    // Scores: 4 key sub-tiles of 16, chained over Dh = 64 (8 WMMA).
    v8f s[4] = {};
#pragma unroll
    for (int nt = 0; nt < 4; ++nt) {
      s[nt] = wmma_f16(qa0, fragB_ld(Kt, nt * 16, LDK, 0), s[nt]);
      s[nt] = wmma_f16(qa1, fragB_ld(Kt, nt * 16, LDK, 32), s[nt]);
    }

    // Online softmax: only the row-max needs a cross-lane reduction.
#pragma unroll
    for (int r = 0; r < 8; ++r) {
      float tmax = fmaxf(fmaxf(s[0][r], s[1][r]), fmaxf(s[2][r], s[3][r]));
#pragma unroll
      for (int msk = 1; msk < 16; msk <<= 1)
        tmax = fmaxf(tmax, __shfl_xor(tmax, msk, 32));
      const float mnew = fmaxf(mrow[r], tmax);
      const float corr = __expf(mrow[r] - mnew);
      mrow[r] = mnew;
      accl[r] *= corr;
#pragma unroll
      for (int dt = 0; dt < 4; ++dt) acc[dt][r] *= corr;
#pragma unroll
      for (int nt = 0; nt < 4; ++nt) {
        const float pv = __expf(s[nt][r] - mnew);
        Pw[(r + 8 * grp) * LDP + nt * 16 + cl] = (half_t)pv;
      }
    }

    // Reload P as A fragments (wave-private LDS; compiler inserts dscnt wait).
    const v16h pa0 = fragA_ld(Pw, 0, LDP, 0);
    const v16h pa1 = fragA_ld(Pw, 0, LDP, 32);

    // Denominator via matrix pipe: accl += P @ ones.
    accl = wmma_f16(pa0, vones, accl);
    accl = wmma_f16(pa1, vones, accl);

    // Output: acc += P @ V  (8 WMMA).
#pragma unroll
    for (int dt = 0; dt < 4; ++dt) {
      acc[dt] = wmma_f16(pa0, fragB_ld(Vt, dt * 16, LDV, 0), acc[dt]);
      acc[dt] = wmma_f16(pa1, fragB_ld(Vt, dt * 16, LDV, 32), acc[dt]);
    }

    __syncthreads();  // protect Kt/Vt before next tile's staging
  }

  // Epilogue: normalize and write f16 values, layout (B*Nq) x (H*Dh).
#pragma unroll
  for (int r = 0; r < 8; ++r) {
    const float inv = 1.0f / accl[r];
    const size_t row = (size_t)(b * Nq + qBase + wave * 16 + r + 8 * grp);
#pragma unroll
    for (int dt = 0; dt < 4; ++dt) {
      valf[row * 1024 + h * 64 + dt * 16 + cl] = (half_t)(acc[dt][r] * inv);
    }
  }
}

// ---------------------------------------------------------------------------
// Launch: Q-proj -> KV-proj -> flash attention -> out-proj (all on `stream`).
// ---------------------------------------------------------------------------
extern "C" void kernel_launch(void* const* d_in, const int* in_sizes, int n_in,
                              void* d_out, int out_size, void* d_ws,
                              size_t ws_size, hipStream_t stream) {
  const float* input_q  = (const float*)d_in[0];
  const float* input_kv = (const float*)d_in[1];
  const float* w_q      = (const float*)d_in[2];
  const float* b_q      = (const float*)d_in[3];
  const float* w_kv     = (const float*)d_in[4];
  const float* b_kv     = (const float*)d_in[5];
  const float* w_out    = (const float*)d_in[6];
  const float* b_out    = (const float*)d_in[7];

  const int B = 4, Nq = 2048, N = 2048;
  const int M = B * Nq;  // 8192 query rows

  // Workspace carve (f16): q 16MB + kv 32MB + values 16MB = 64MB.
  half_t* qf   = (half_t*)d_ws;
  half_t* kvf  = qf + (size_t)M * 1024;
  half_t* valf = kvf + (size_t)M * 2048;

  const dim3 blk(256);

  // 1) q = (x @ w_q^T + b_q) * (1/sqrt(Dh))  [softmax scale folded into q]
  gemm_bias_kernel<float, false>
      <<<dim3(1024 / 128, M / 128), blk, 0, stream>>>(input_q, w_q, b_q, qf, M,
                                                      1024, 1024, 0.125f);
  // 2) kv = x_kv @ w_kv^T + b_kv
  gemm_bias_kernel<float, false>
      <<<dim3(2048 / 128, M / 128), blk, 0, stream>>>(input_kv, w_kv, b_kv,
                                                      kvf, M, 2048, 1024, 1.0f);
  // 3) flash attention per (b, h) and 128-query blocks
  flash_attn_kernel<<<dim3(Nq / 128, B * 16), blk, 0, stream>>>(qf, kvf, valf,
                                                                B, Nq, N);
  // 4) out = values @ w_out^T + b_out  (fp32 output)
  gemm_bias_kernel<half_t, true>
      <<<dim3(1024 / 128, M / 128), blk, 0, stream>>>(valf, w_out, b_out,
                                                      d_out, M, 1024, 1024,
                                                      1.0f);
}
